// DeformableAlignment_59785944760673
// MI455X (gfx1250) — matrix operand order repaired and checked by reading, not compiled, not run on hardware
//
#include <hip/hip_runtime.h>

// Problem constants
#define B_   2
#define C_   64
#define H_   256
#define W_   256
#define DG_  8
#define KK_  9
#define CG_  8
#define C2   128     // 2*C (concat channels)
#define KC   1152    // im2col K for offset/mask conv: k = tap*128 + ci
#define NP   224     // padded N for offset+mask conv (144+72=216 -> 224)
#define NREAL 216
#define NOFF 144
#define K2   576     // final gemm K = DG*CG*KK
#define OMS  224     // per-pixel stride (floats) of offset/mask buffer

typedef __attribute__((ext_vector_type(16))) __bf16 v16bf;
typedef __attribute__((ext_vector_type(8)))  __bf16 v8bf;
typedef __attribute__((ext_vector_type(8)))  float  v8f;

union V16U { v16bf v; v8bf h[2]; };

// ---- CDNA5 async copy (global -> LDS, ASYNCcnt-tracked) with fallback ------
#if defined(__has_builtin)
#if __has_builtin(__builtin_amdgcn_global_load_async_to_lds_b128) && \
    __has_builtin(__builtin_amdgcn_s_wait_asynccnt)
#define USE_ASYNC 1
#endif
#endif
#ifndef USE_ASYNC
#define USE_ASYNC 0
#endif

#if USE_ASYNC
typedef int v4i __attribute__((vector_size(4 * sizeof(int))));
typedef __attribute__((address_space(1))) v4i gv4i;   // global int4
typedef __attribute__((address_space(3))) v4i lv4i;   // LDS int4
#define ASYNC_WAIT() __builtin_amdgcn_s_wait_asynccnt(0)
#else
#define ASYNC_WAIT()
#endif

__device__ __forceinline__ int iclamp(int v, int lo, int hi) {
    return v < lo ? lo : (v > hi ? hi : v);
}

// ---------------------------------------------------------------------------
// Pack concat([ref, neighbour]) into bf16, channel-contiguous per pixel:
// X[((b*H + y)*W + x)*128 + c]
// ---------------------------------------------------------------------------
__global__ void pack_input_kernel(const float* __restrict__ ref,
                                  const float* __restrict__ nbr,
                                  __bf16* __restrict__ X) {
    const size_t e   = (size_t)blockIdx.x * blockDim.x + threadIdx.x; // B*H*W*128
    const int    c   = (int)(e & (C2 - 1));
    const size_t pix = e >> 7;               // b*H*W + sp
    const size_t b   = pix / (size_t)(H_ * W_);
    const size_t sp  = pix % (size_t)(H_ * W_);
    const float v = (c < C_)
        ? ref[(b * C_ + c)        * (size_t)(H_ * W_) + sp]
        : nbr[(b * C_ + (c - C_)) * (size_t)(H_ * W_) + sp];
    X[e] = (__bf16)v;
}

// ---------------------------------------------------------------------------
// Pack offset conv (144 rows) + mask conv (72 rows) weights into a single
// bf16 B-matrix  Wcm[n][k],  k = tap*128 + ci,  n padded to 224.
// off_w / mask_w layout: (n, 128, 3, 3) -> idx = (n*128 + ci)*9 + tap
// ---------------------------------------------------------------------------
__global__ void pack_wcm_kernel(const float* __restrict__ off_w,
                                const float* __restrict__ mask_w,
                                __bf16* __restrict__ Wcm) {
    const int e = blockIdx.x * blockDim.x + threadIdx.x;   // NP*KC
    const int n = e / KC;
    const int k = e - n * KC;
    const int tap = k >> 7;        // k / 128
    const int ci  = k & 127;
    float v = 0.0f;
    if (n < NOFF)       v = off_w [((size_t)n * C2 + ci) * KK_ + tap];
    else if (n < NREAL) v = mask_w[((size_t)(n - NOFF) * C2 + ci) * KK_ + tap];
    Wcm[e] = (__bf16)v;
}

// Final conv weight (C, C, 3, 3) flat already matches Wf[o][k], k = ci*9+kk.
__global__ void pack_wf_kernel(const float* __restrict__ w,
                               __bf16* __restrict__ Wf) {
    const int e = blockIdx.x * blockDim.x + threadIdx.x;   // 64*576
    Wf[e] = (__bf16)w[e];
}

// ---------------------------------------------------------------------------
// Cooperative stage of one 32-K x 224-N weight panel (14336 B) into LDS.
// 128 threads x 7 transfers x 16 B.
// ---------------------------------------------------------------------------
__device__ __forceinline__ void stage_panel(const __bf16* __restrict__ Wcm,
                                            __bf16* dst, int kc, int t) {
#pragma unroll
    for (int i = 0; i < 7; ++i) {
        const int j   = t + i * 128;     // 0..895
        const int row = j >> 2;          // 0..223
        const int seg = j & 3;           // 16B segment within 64B row
        const __bf16* src = Wcm + (size_t)row * KC + kc * 32 + seg * 8;
        __bf16* d = dst + row * 32 + seg * 8;
#if USE_ASYNC
        __builtin_amdgcn_global_load_async_to_lds_b128(
            (gv4i*)src, (lv4i*)d, 0, 0);
#else
        *(v8bf*)d = *(const v8bf*)src;
#endif
    }
}

// ---------------------------------------------------------------------------
// Implicit-GEMM 3x3 conv producing offsets (raw) + mask logits (raw):
// per wave: M = 16 pixels, N = 224 (14 tiles), K = 1152 (36 chunks of 32).
// B panel double-buffered in LDS, shared by the 4 waves of the block.
// Output: OM[((b*H+y)*W + x)*224 + n]  (fp32, pixel-major)
// ---------------------------------------------------------------------------
__global__ void __launch_bounds__(128)
conv_offmask_kernel(const __bf16* __restrict__ X,
                    const __bf16* __restrict__ Wcm,
                    const float* __restrict__ off_b,
                    const float* __restrict__ mask_b,
                    float* __restrict__ OM) {
    __shared__ __align__(16) __bf16 Bpan[2][NP * 32];   // 2 x 14336 B

    const int t    = (int)threadIdx.x;
    const int lane = t & 31;
    const int wave = t >> 5;
    const int tile = blockIdx.x * 4 + wave;          // B*H*(W/16) = 8192 tiles
    const int b    = tile / (H_ * (W_ / 16));
    const int rem  = tile % (H_ * (W_ / 16));
    const int y    = rem / (W_ / 16);
    const int x0   = (rem % (W_ / 16)) * 16;

    const int m  = lane & 15;       // pixel within tile / A-matrix row
    const int hv = lane >> 4;       // lane-half selector (ISA A/B layouts)
    const int px = x0 + m;

    v8f acc[14];
#pragma unroll
    for (int nt = 0; nt < 14; ++nt)
#pragma unroll
        for (int q = 0; q < 8; ++q) acc[nt][q] = 0.0f;

    stage_panel(Wcm, Bpan[0], 0, t);
    ASYNC_WAIT();
    __syncthreads();

#pragma unroll 1
    for (int kc = 0; kc < 36; ++kc) {               // K chunks of 32
        const int cur = kc & 1;
        if (kc < 35) stage_panel(Wcm, Bpan[cur ^ 1], kc + 1, t);

        // ---- A fragment (16x32 bf16): lane holds K=hv*16 + i
        const int tap = kc >> 2;                    // 9 taps, 4 chunks each
        const int cb  = (kc & 3) * 32;              // channel base within tap
        const int ys  = y  + tap / 3 - 1;
        const int xs  = px + tap % 3 - 1;
        V16U ua;
#pragma unroll
        for (int q = 0; q < 8; ++q) { ua.h[0][q] = (__bf16)0.0f; ua.h[1][q] = (__bf16)0.0f; }
        if (ys >= 0 && ys < H_ && xs >= 0 && xs < W_) {
            const __bf16* p = X + (size_t)((b * H_ + ys) * W_ + xs) * C2;
            ua.h[0] = *(const v8bf*)(p + cb + hv * 16);
            ua.h[1] = *(const v8bf*)(p + cb + hv * 16 + 8);
        }
        const v16bf a = ua.v;

        // ---- B fragments from LDS: lane holds N=lane&15, K=hv*16 + i
        const __bf16* pb = &Bpan[cur][(lane & 15) * 32 + hv * 16];
        v16bf bfr[7];
#pragma unroll
        for (int u = 0; u < 7; ++u) bfr[u] = *(const v16bf*)(pb + u * (16 * 32));
#pragma unroll
        for (int u = 0; u < 7; ++u)
            acc[u] = __builtin_amdgcn_wmma_f32_16x16x32_bf16(
                false, a, false, bfr[u], (short)0, acc[u], false, false);
#pragma unroll
        for (int u = 0; u < 7; ++u) bfr[u] = *(const v16bf*)(pb + (7 + u) * (16 * 32));
#pragma unroll
        for (int u = 0; u < 7; ++u)
            acc[7 + u] = __builtin_amdgcn_wmma_f32_16x16x32_bf16(
                false, a, false, bfr[u], (short)0, acc[7 + u], false, false);

        ASYNC_WAIT();
        __syncthreads();
    }

    // D layout: vgpr r, lane -> M = r + 8*hv (pixel), N = lane&15 (channel)
    const size_t pixrow = (size_t)(b * H_ + y) * W_;
#pragma unroll
    for (int nt = 0; nt < 14; ++nt) {
        const int n = nt * 16 + (lane & 15);
        float bz = 0.0f;
        if (n < NOFF)       bz = off_b[n];
        else if (n < NREAL) bz = mask_b[n - NOFF];
        if (n < NREAL) {
#pragma unroll
            for (int rr = 0; rr < 8; ++rr) {
                const int M = rr + hv * 8;
                OM[(pixrow + x0 + M) * OMS + n] = acc[nt][rr] + bz;
            }
        }
    }
}

// ---------------------------------------------------------------------------
// Fused deformable sampling + mask + final GEMM.
// One wave per 16-pixel tile:
//  phase 1: bilinear-sample neighbour at offset positions, apply sigmoid(mask),
//           write bf16 A-tile [16 x 576] to LDS.
//  phase 2: WMMA  out[16 x 64] = A x Wf^T, add bias, store NCHW fp32.
// ---------------------------------------------------------------------------
__global__ void __launch_bounds__(32)
sample_gemm_kernel(const float* __restrict__ neigh,
                   const float* __restrict__ OM,
                   const __bf16* __restrict__ Wf,
                   const float* __restrict__ bias,
                   float* __restrict__ out) {
    __shared__ __align__(16) __bf16 At[16 * K2];   // 18 KB

    const int lane = threadIdx.x & 31;
    const int tile = blockIdx.x;                   // 8192 tiles
    const int b    = tile / (H_ * (W_ / 16));
    const int rem  = tile % (H_ * (W_ / 16));
    const int y    = rem / (W_ / 16);
    const int x0   = (rem % (W_ / 16)) * 16;
    const int m    = lane & 15;
    const int hv   = lane >> 4;
    const int px   = x0 + m;

    const float* om = OM + (size_t)((b * H_ + y) * W_ + px) * OMS;

    // phase 1: each lane covers pixel m, channels [hv*4, hv*4+4) of its group
    for (int g = 0; g < DG_; ++g) {
#pragma unroll 1
        for (int kk = 0; kk < KK_; ++kk) {
            const int chd = (g * KK_ + kk) * 2;
            const float dy   = om[chd];
            const float dx   = om[chd + 1];
            const float mraw = om[NOFF + g * KK_ + kk];
            const float mval = 1.0f / (1.0f + __expf(-mraw));   // v_exp_f32

            const float ysf = (float)(y  - 1 + kk / 3) + dy;
            const float xsf = (float)(px - 1 + kk % 3) + dx;
            const float y0f = floorf(ysf), x0f = floorf(xsf);
            const float wy = ysf - y0f, wx = xsf - x0f;
            const int iy0 = (int)y0f, ix0 = (int)x0f;
            const int iy1 = iy0 + 1,  ix1 = ix0 + 1;

            const float v00 = (iy0 >= 0 && iy0 < H_ && ix0 >= 0 && ix0 < W_) ? 1.0f : 0.0f;
            const float v01 = (iy0 >= 0 && iy0 < H_ && ix1 >= 0 && ix1 < W_) ? 1.0f : 0.0f;
            const float v10 = (iy1 >= 0 && iy1 < H_ && ix0 >= 0 && ix0 < W_) ? 1.0f : 0.0f;
            const float v11 = (iy1 >= 0 && iy1 < H_ && ix1 >= 0 && ix1 < W_) ? 1.0f : 0.0f;
            const float w00 = (1.0f - wy) * (1.0f - wx) * v00;
            const float w01 = (1.0f - wy) * wx          * v01;
            const float w10 = wy          * (1.0f - wx) * v10;
            const float w11 = wy          * wx          * v11;
            const int cy0 = iclamp(iy0, 0, H_ - 1), cx0 = iclamp(ix0, 0, W_ - 1);
            const int cy1 = iclamp(iy1, 0, H_ - 1), cx1 = iclamp(ix1, 0, W_ - 1);

#pragma unroll
            for (int c = 0; c < 4; ++c) {
                const int cc = g * CG_ + hv * 4 + c;
                const float* img = neigh + (size_t)(b * C_ + cc) * (H_ * W_);
                const float v = w00 * img[cy0 * W_ + cx0] + w01 * img[cy0 * W_ + cx1]
                              + w10 * img[cy1 * W_ + cx0] + w11 * img[cy1 * W_ + cx1];
                At[m * K2 + g * (CG_ * KK_) + (hv * 4 + c) * KK_ + kk] = (__bf16)(v * mval);
            }
        }
    }
    __syncthreads();

    // phase 2: 16x64 = (16x576) x (576x64) via 18 K-chunks x 4 N-tiles
    v8f acc[4];
#pragma unroll
    for (int nt = 0; nt < 4; ++nt)
#pragma unroll
        for (int q = 0; q < 8; ++q) acc[nt][q] = 0.0f;

#pragma unroll 1
    for (int kc = 0; kc < 18; ++kc) {
        V16U ua;
        ua.h[0] = *(const v8bf*)(At + m * K2 + kc * 32 + hv * 16);       // ds_load_b128
        ua.h[1] = *(const v8bf*)(At + m * K2 + kc * 32 + hv * 16 + 8);
        const v16bf a = ua.v;
        const __bf16* wb = Wf + (size_t)(lane & 15) * K2 + kc * 32 + hv * 16;
        v16bf bfr[4];
#pragma unroll
        for (int nt = 0; nt < 4; ++nt)
            bfr[nt] = *(const v16bf*)(wb + (size_t)nt * 16 * K2);
#pragma unroll
        for (int nt = 0; nt < 4; ++nt)
            acc[nt] = __builtin_amdgcn_wmma_f32_16x16x32_bf16(
                false, a, false, bfr[nt], (short)0, acc[nt], false, false);
    }

#pragma unroll
    for (int nt = 0; nt < 4; ++nt) {
        const int n  = nt * 16 + (lane & 15);
        const float bv = bias[n];
#pragma unroll
        for (int rr = 0; rr < 8; ++rr) {
            const int M = rr + hv * 8;
            out[((size_t)(b * C_ + n) * H_ + y) * W_ + x0 + M] = acc[nt][rr] + bv;
        }
    }
}

// ---------------------------------------------------------------------------
extern "C" void kernel_launch(void* const* d_in, const int* in_sizes, int n_in,
                              void* d_out, int out_size, void* d_ws, size_t ws_size,
                              hipStream_t stream) {
    const float* ref    = (const float*)d_in[0];
    const float* nbr    = (const float*)d_in[1];
    const float* off_w  = (const float*)d_in[2];
    const float* off_b  = (const float*)d_in[3];
    const float* mask_w = (const float*)d_in[4];
    const float* mask_b = (const float*)d_in[5];
    const float* weight = (const float*)d_in[6];
    const float* bias   = (const float*)d_in[7];
    float* out = (float*)d_out;

    char* ws = (char*)d_ws;
    size_t off = 0;
    auto take = [&](size_t bytes) {
        char* p = ws + off;
        off = (off + bytes + 255) & ~(size_t)255;
        return p;
    };
    __bf16* X   = (__bf16*)take((size_t)B_ * H_ * W_ * C2 * 2);   // 33.6 MB
    __bf16* Wcm = (__bf16*)take((size_t)NP * KC * 2);             // 0.5 MB
    __bf16* Wf  = (__bf16*)take((size_t)C_ * K2 * 2);             // 72 KB
    float*  OM  = (float*) take((size_t)B_ * H_ * W_ * OMS * 4);  // 117 MB

    hipLaunchKernelGGL(pack_input_kernel, dim3((B_ * H_ * W_ * C2) / 256), dim3(256),
                       0, stream, ref, nbr, X);
    hipLaunchKernelGGL(pack_wcm_kernel, dim3((NP * KC) / 256), dim3(256),
                       0, stream, off_w, mask_w, Wcm);
    hipLaunchKernelGGL(pack_wf_kernel, dim3((C_ * K2) / 256), dim3(256),
                       0, stream, weight, Wf);
    hipLaunchKernelGGL(conv_offmask_kernel, dim3((B_ * H_ * (W_ / 16)) / 4), dim3(128),
                       0, stream, X, Wcm, off_b, mask_b, OM);
    hipLaunchKernelGGL(sample_gemm_kernel, dim3(B_ * H_ * (W_ / 16)), dim3(32),
                       0, stream, nbr, OM, Wf, bias, out);
}
